// RC_36137854828559
// MI455X (gfx1250) — matrix operand order repaired and checked by reading, not compiled, not run on hardware
//
#include <hip/hip_runtime.h>
#include <hip/hip_bf16.h>

typedef __attribute__((ext_vector_type(16))) __bf16 v16bf;
typedef __attribute__((ext_vector_type(8)))  __bf16 v8bf;
typedef __attribute__((ext_vector_type(8)))  float  v8f;

#define BM 128          // block tile M
#define BNT 256         // block tile N
#define BK 32           // k-chunk
#define LDS_STRIDE 40   // 32 k-elements + 8 pad (80B row stride, conflict-free)

#define B_DIM 1024
#define IN_DIM 256
#define R_DIM 2048
#define OUT_DIM 256
#define N_STEPS 100

// CDNA5 async global->LDS copy path (ASYNCcnt), guarded so missing builtins
// fall back to the synchronous global->VGPR->ds_store path.
#if __has_builtin(__builtin_amdgcn_global_load_async_to_lds_b128) && \
    __has_builtin(__builtin_amdgcn_s_wait_asynccnt)
#define RC_ASYNC 1
#define AS1 __attribute__((address_space(1)))
#define AS3 __attribute__((address_space(3)))
typedef int rc_v4i __attribute__((vector_size(16)));
#define RC_ASYNC_CP16(gsrc, ldst)                                              \
  __builtin_amdgcn_global_load_async_to_lds_b128((AS1 rc_v4i*)(gsrc),          \
                                                 (AS3 rc_v4i*)(ldst), 0, 0)
#else
#define RC_ASYNC 0
#endif

// ---------------- f32 -> bf16 conversion ----------------
__global__ void rc_cvt_f32_bf16(const float* __restrict__ in,
                                __bf16* __restrict__ out, int n) {
  int i = blockIdx.x * blockDim.x + threadIdx.x;
  int stride = gridDim.x * blockDim.x;
  for (; i < n; i += stride) out[i] = (__bf16)in[i];
}

// ---------------- bf16 WMMA GEMM: C = [tanh](A @ W^T [+ bias]) ----------------
// A: [M x K] row-major bf16 ; W: [N x K] row-major bf16 (nn.Linear layout)
// M divisible by 128, N divisible by 256, K divisible by 32.
union FragAB { v16bf v; v8bf h[2]; };

template <bool TANH, bool HAS_BIAS, bool OUT_BF16>
__global__ __launch_bounds__(256) void rc_gemm_bf16(
    const __bf16* __restrict__ A,
    const __bf16* __restrict__ W,
    const float* __restrict__ bias,
    void* __restrict__ Cout,
    int M, int N, int K) {
  __shared__ __bf16 sA[2][BM * LDS_STRIDE];
  __shared__ __bf16 sW[2][BNT * LDS_STRIDE];

  const int tid   = threadIdx.x;
  const int lane  = tid & 31;
  const int wave  = tid >> 5;
  const int waveM = wave & 1;   // 2 waves along M -> 64 rows each
  const int waveN = wave >> 1;  // 4 waves along N -> 64 cols each
  const int l16   = lane & 15;
  const int half  = lane >> 4;  // 0: lanes 0-15, 1: lanes 16-31

  const int blockM = blockIdx.y * BM;
  const int blockN = blockIdx.x * BNT;

  // loader: 16B chunks; thread t owns (row = t>>2 (+64j), col = (t&3)*8) of each tile
  const int ldRow = tid >> 2;        // 0..63
  const int ldCol = (tid & 3) * 8;   // 0,8,16,24

  const __bf16* gA = A + (size_t)(blockM + ldRow) * K + ldCol;
  const __bf16* gW = W + (size_t)(blockN + ldRow) * K + ldCol;

  v8f acc[4][4];
#pragma unroll
  for (int i = 0; i < 4; i++)
#pragma unroll
    for (int j = 0; j < 4; j++) acc[i][j] = v8f{};

#if RC_ASYNC
  // prologue: stage kc = 0 into buffer 0
#pragma unroll
  for (int j = 0; j < 2; j++)
    RC_ASYNC_CP16(gA + (size_t)(64 * j) * K,
                  &sA[0][(ldRow + 64 * j) * LDS_STRIDE + ldCol]);
#pragma unroll
  for (int j = 0; j < 4; j++)
    RC_ASYNC_CP16(gW + (size_t)(64 * j) * K,
                  &sW[0][(ldRow + 64 * j) * LDS_STRIDE + ldCol]);
#endif

  int buf = 0;
  for (int kc = 0; kc < K; kc += BK) {
#if RC_ASYNC
    __builtin_amdgcn_s_wait_asynccnt(0);  // own async loads into sA/sW[buf] done
    __syncthreads();                      // everyone's loads visible; prev reads done
    if (kc + BK < K) {                    // stage next chunk while computing this one
      const int nb = buf ^ 1;
#pragma unroll
      for (int j = 0; j < 2; j++)
        RC_ASYNC_CP16(gA + (size_t)(64 * j) * K + kc + BK,
                      &sA[nb][(ldRow + 64 * j) * LDS_STRIDE + ldCol]);
#pragma unroll
      for (int j = 0; j < 4; j++)
        RC_ASYNC_CP16(gW + (size_t)(64 * j) * K + kc + BK,
                      &sW[nb][(ldRow + 64 * j) * LDS_STRIDE + ldCol]);
    }
#else
    // synchronous fallback: global -> VGPR -> LDS (alternating buffers, 2 barriers)
    {
      v8bf ar[2], wr[4];
#pragma unroll
      for (int j = 0; j < 2; j++)
        ar[j] = *(const v8bf*)(gA + (size_t)(64 * j) * K + kc);
#pragma unroll
      for (int j = 0; j < 4; j++)
        wr[j] = *(const v8bf*)(gW + (size_t)(64 * j) * K + kc);
      if (kc + BK < K) {  // global_prefetch_b8 for next chunk
        __builtin_prefetch(gA + kc + BK, 0, 1);
        __builtin_prefetch(gW + (size_t)192 * K + kc + BK, 0, 1);
      }
#pragma unroll
      for (int j = 0; j < 2; j++)
        *(v8bf*)&sA[buf][(ldRow + 64 * j) * LDS_STRIDE + ldCol] = ar[j];
#pragma unroll
      for (int j = 0; j < 4; j++)
        *(v8bf*)&sW[buf][(ldRow + 64 * j) * LDS_STRIDE + ldCol] = wr[j];
    }
    __syncthreads();
#endif

    // ---- fragments (layouts per CDNA5 ISA 16-bit A/B tables) ----
    FragAB afr[4], bfr[4];
#pragma unroll
    for (int fm = 0; fm < 4; fm++) {
      const __bf16* base = &sA[buf][(waveM * 64 + fm * 16 + l16) * LDS_STRIDE];
      const int koff = half * 8;  // lane<16: K 0..7/16..23 ; lane>=16: K 8..15/24..31
      afr[fm].h[0] = *(const v8bf*)(base + koff);
      afr[fm].h[1] = *(const v8bf*)(base + 16 + koff);
    }
#pragma unroll
    for (int fn = 0; fn < 4; fn++) {
      const __bf16* base = &sW[buf][(waveN * 64 + fn * 16 + l16) * LDS_STRIDE];
      const int koff = half * 8;  // lane<16: K 0..7/16..23 ; lane>=16: K 8..15/24..31
      bfr[fn].h[0] = *(const v8bf*)(base + koff);
      bfr[fn].h[1] = *(const v8bf*)(base + 16 + koff);
    }

#pragma unroll
    for (int fm = 0; fm < 4; fm++)
#pragma unroll
      for (int fn = 0; fn < 4; fn++)
        acc[fm][fn] = __builtin_amdgcn_wmma_f32_16x16x32_bf16(
            /*neg_a=*/false, afr[fm].v, /*neg_b=*/false, bfr[fn].v,
            /*c_mod=*/(short)0, acc[fm][fn],
            /*reuse_a=*/false, /*reuse_b=*/false);

#if !RC_ASYNC
    __syncthreads();  // async path's next-iter barrier covers this
#endif
    buf ^= 1;
  }

  // ---- epilogue: C VGPR i -> (M = i + half*8, N = l16) ----
  const int mBase = blockM + waveM * 64;
  const int nBase = blockN + waveN * 64;
#pragma unroll
  for (int fm = 0; fm < 4; fm++) {
#pragma unroll
    for (int fn = 0; fn < 4; fn++) {
      const int n = nBase + fn * 16 + l16;
      const float bv = HAS_BIAS ? bias[n] : 0.0f;
#pragma unroll
      for (int i = 0; i < 8; i++) {
        const int m = mBase + fm * 16 + half * 8 + i;
        float v = acc[fm][fn][i] + bv;
        if (TANH) v = tanhf(v);
        if (OUT_BF16)
          ((__bf16*)Cout)[(size_t)m * N + n] = (__bf16)v;
        else
          ((float*)Cout)[(size_t)m * N + n] = v;
      }
    }
  }
}

static inline int cvt_grid(int n) {
  int g = (n + 255) / 256;
  return g > 4096 ? 4096 : g;
}

extern "C" void kernel_launch(void* const* d_in, const int* in_sizes, int n_in,
                              void* d_out, int out_size, void* d_ws, size_t ws_size,
                              hipStream_t stream) {
  const float* x      = (const float*)d_in[0];  // [B, IN]
  const float* Win_w  = (const float*)d_in[1];  // [R, IN]
  const float* Win_b  = (const float*)d_in[2];  // [R]
  const float* Wres_w = (const float*)d_in[3];  // [R, R]
  const float* Wout_w = (const float*)d_in[4];  // [OUT, R]
  const float* Wout_b = (const float*)d_in[5];  // [OUT]

  char* ws = (char*)d_ws;
  __bf16* x_bf    = (__bf16*)ws; ws += (size_t)B_DIM * IN_DIM * 2;
  __bf16* Win_bf  = (__bf16*)ws; ws += (size_t)R_DIM * IN_DIM * 2;
  __bf16* Wres_bf = (__bf16*)ws; ws += (size_t)R_DIM * R_DIM * 2;
  __bf16* Wout_bf = (__bf16*)ws; ws += (size_t)OUT_DIM * R_DIM * 2;
  __bf16* h0      = (__bf16*)ws; ws += (size_t)B_DIM * R_DIM * 2;
  __bf16* h1      = (__bf16*)ws; ws += (size_t)B_DIM * R_DIM * 2;

  // one-time (per launch) f32 -> bf16 conversions
  {
    int n;
    n = B_DIM * IN_DIM;   rc_cvt_f32_bf16<<<cvt_grid(n), 256, 0, stream>>>(x, x_bf, n);
    n = R_DIM * IN_DIM;   rc_cvt_f32_bf16<<<cvt_grid(n), 256, 0, stream>>>(Win_w, Win_bf, n);
    n = R_DIM * R_DIM;    rc_cvt_f32_bf16<<<cvt_grid(n), 256, 0, stream>>>(Wres_w, Wres_bf, n);
    n = OUT_DIM * R_DIM;  rc_cvt_f32_bf16<<<cvt_grid(n), 256, 0, stream>>>(Wout_w, Wout_bf, n);
  }

  const dim3 blk(256);

  // h0 = tanh(x @ Win^T + Win_b)   [M=B, N=R, K=IN]
  rc_gemm_bf16<true, true, true>
      <<<dim3(R_DIM / BNT, B_DIM / BM), blk, 0, stream>>>(
          x_bf, Win_bf, Win_b, (void*)h0, B_DIM, R_DIM, IN_DIM);

  // 100 x: h = tanh(h @ Wres^T)    [M=B, N=R, K=R]
  __bf16* hc = h0;
  __bf16* hn = h1;
  for (int s = 0; s < N_STEPS; s++) {
    rc_gemm_bf16<true, false, true>
        <<<dim3(R_DIM / BNT, B_DIM / BM), blk, 0, stream>>>(
            hc, Wres_bf, nullptr, (void*)hn, B_DIM, R_DIM, R_DIM);
    __bf16* t = hc; hc = hn; hn = t;
  }

  // out = h @ Wout^T + Wout_b      [M=B, N=OUT, K=R], f32 output
  rc_gemm_bf16<false, true, false>
      <<<dim3(OUT_DIM / BNT, B_DIM / BM), blk, 0, stream>>>(
          hc, Wout_bf, Wout_b, d_out, B_DIM, OUT_DIM, R_DIM);
}
